// SyllableRNN_8761733284404
// MI455X (gfx1250) — matrix operand order repaired and checked by reading, not compiled, not run on hardware
//
#include <hip/hip_runtime.h>
#include <hip/hip_bf16.h>

typedef __attribute__((ext_vector_type(16))) _Float16 v16h;
typedef __attribute__((ext_vector_type(8)))  float    v8f;

#define B_SZ   1024
#define T_SZ   128
#define E_SZ   256
#define H_SZ   1024
#define TAGS   8
#define VOCAB  128
#define HPAD   1032   /* 1024 + 8 halves pad -> LDS bank stride 4 */
#define BT     16     /* batch rows per block (WMMA M) */
#define TPB    256
#define NWAVE  8

/* workspace layout (in _Float16 units) */
#define EMB16_OFF  0
#define WHH_OFF    (VOCAB * E_SZ)                 /* 32768   */
#define WIH_OFF    (WHH_OFF + H_SZ * H_SZ)        /* +1M     */
#define WOUT_OFF   (WIH_OFF + H_SZ * E_SZ)        /* +256K   */

/* ---------------- prep kernels: f32 -> f16 + fragment swizzle ---------------- */

__global__ void prep_emb(const float* __restrict__ emb, _Float16* __restrict__ dst) {
    int i = blockIdx.x * blockDim.x + threadIdx.x;
    if (i < VOCAB * E_SZ) dst[i] = (_Float16)emb[i];
}

/* B-fragment swizzle: dst[((ntile*KK + kk)*32 + lane)*16 + j] =
   W[ntile*16 + lane%16][kk*32 + (lane/16)*16 + j]  (lane holds 32B contiguous) */
__global__ void prep_whh(const float* __restrict__ W, _Float16* __restrict__ dst) {
    int idx = blockIdx.x * blockDim.x + threadIdx.x;
    if (idx >= H_SZ * H_SZ) return;
    int j  = idx & 15;
    int ln = (idx >> 4) & 31;
    int kk = (idx >> 9) & 31;
    int nt = idx >> 14;
    int row = nt * 16 + (ln & 15);
    int col = kk * 32 + (ln >> 4) * 16 + j;
    dst[idx] = (_Float16)W[row * H_SZ + col];
}

__global__ void prep_wih(const float* __restrict__ W, _Float16* __restrict__ dst) {
    int idx = blockIdx.x * blockDim.x + threadIdx.x;
    if (idx >= H_SZ * E_SZ) return;
    int j  = idx & 15;
    int ln = (idx >> 4) & 31;
    int kk = (idx >> 9) & 7;
    int nt = idx >> 12;
    int row = nt * 16 + (ln & 15);
    int col = kk * 32 + (ln >> 4) * 16 + j;
    dst[idx] = (_Float16)W[row * E_SZ + col];
}

__global__ void prep_wout(const float* __restrict__ W, _Float16* __restrict__ dst) {
    int idx = blockIdx.x * blockDim.x + threadIdx.x;
    if (idx >= 32 * 32 * 16) return;
    int j  = idx & 15;
    int ln = (idx >> 4) & 31;
    int kk = idx >> 9;
    int row = ln & 15;
    int col = kk * 32 + (ln >> 4) * 16 + j;
    dst[idx] = (row < TAGS) ? (_Float16)W[row * H_SZ + col] : (_Float16)0.0f;
}

/* ---------------- persistent per-batch-tile RNN kernel ---------------- */

union AF { v16h v; uint4 q[2]; };

__device__ __forceinline__ void loadA(AF& a, const _Float16* abase, int kk, int half) {
    a.q[0] = *(const uint4*)(abase + kk * 32 + half * 16);
    a.q[1] = *(const uint4*)(abase + kk * 32 + half * 16 + 8);
}

__global__ __launch_bounds__(TPB) void rnn_kernel(
    const int*      __restrict__ gid,
    const _Float16* __restrict__ emb16,
    const uint4*    __restrict__ whh,   /* swizzled B fragments */
    const uint4*    __restrict__ wih,
    const uint4*    __restrict__ wout,
    const float*    __restrict__ b_ih,
    const float*    __restrict__ b_hh,
    const float*    __restrict__ b_out,
    float*          __restrict__ out)
{
    __shared__ _Float16 h_lds[BT * HPAD];
    __shared__ float    bias_lds[H_SZ];
    __shared__ float    outscr[NWAVE * 256];
    __shared__ int      ids_all[BT * T_SZ];
    __shared__ int      slen[BT];

    const int tid  = threadIdx.x;
    const int lane = tid & 31;
    const int wave = tid >> 5;
    const int r16  = lane & 15;
    const int half = lane >> 4;
    const int b0   = blockIdx.x * BT;
    const _Float16* abase = h_lds + r16 * HPAD;

    for (int i = tid; i < BT * HPAD; i += TPB) h_lds[i] = (_Float16)0.0f;
    for (int i = tid; i < H_SZ;      i += TPB) bias_lds[i] = b_ih[i] + b_hh[i];
    for (int i = tid; i < BT * T_SZ; i += TPB)
        ids_all[i] = gid[(size_t)b0 * T_SZ + i];        /* [row][t], rows contiguous */
    if (tid < BT) {
        int c = 0;
        const int* g = gid + (size_t)(b0 + tid) * T_SZ;
        for (int t = 0; t < T_SZ; ++t) c += (g[t] != 0);
        slen[tid] = c;
    }
    __syncthreads();

    /* per-wave constant B base pointers */
    const uint4* whh_base  = whh  + ((size_t)wave * 256 * 32 + lane) * 2; /* +kk*64, +j*2048 */
    const uint4* wih_base  = wih  + ((size_t)wave * 64  * 32 + lane) * 2; /* +kk*64, +j*512  */
    const uint4* wout_base = wout + ((size_t)wave * 4   * 32 + lane) * 2; /* +k4*64          */

    for (int t = 0; t < T_SZ; ++t) {
        v8f acc[8] = {};

        /* ---- recurrent GEMM: Z += h_old @ W_hh^T, K = 1024 (pipelined) ---- */
        {
            AF a0;
            uint4 bb[8][2];
            loadA(a0, abase, 0, half);
#pragma unroll
            for (int j = 0; j < 8; ++j) {
                const uint4* p = whh_base + (size_t)j * 2048;
                bb[j][0] = p[0]; bb[j][1] = p[1];
            }
#pragma unroll 2
            for (int kk = 0; kk < H_SZ / 32; ++kk) {
                const int kn = (kk + 1) & 31;          /* wrap: harmless dummy prefetch */
                AF an;
                uint4 bn[8][2];
                loadA(an, abase, kn, half);
                const uint4* bpn = whh_base + (size_t)kn * 64;
#pragma unroll
                for (int j = 0; j < 8; ++j) {
                    const uint4* p = bpn + (size_t)j * 2048;
                    bn[j][0] = p[0]; bn[j][1] = p[1];
                }
#pragma unroll
                for (int j = 0; j < 8; ++j) {
                    AF b; b.q[0] = bb[j][0]; b.q[1] = bb[j][1];
                    acc[j] = __builtin_amdgcn_wmma_f32_16x16x32_f16(
                        false, a0.v, false, b.v, (short)0, acc[j], false, false);
                }
                a0 = an;
#pragma unroll
                for (int j = 0; j < 8; ++j) { bb[j][0] = bn[j][0]; bb[j][1] = bn[j][1]; }
            }
        }

        /* ---- fused input GEMM: Z += emb[id_t] @ W_ih^T, K = 256 (pipelined) ---- */
        {
            const _Float16* erow = emb16 + (size_t)ids_all[r16 * T_SZ + t] * E_SZ;
            AF a0;
            uint4 bb[8][2];
            a0.q[0] = *(const uint4*)(erow + half * 16);
            a0.q[1] = *(const uint4*)(erow + half * 16 + 8);
#pragma unroll
            for (int j = 0; j < 8; ++j) {
                const uint4* p = wih_base + (size_t)j * 512;
                bb[j][0] = p[0]; bb[j][1] = p[1];
            }
#pragma unroll 2
            for (int kk = 0; kk < E_SZ / 32; ++kk) {
                const int kn = (kk + 1) & 7;
                AF an;
                uint4 bn[8][2];
                an.q[0] = *(const uint4*)(erow + kn * 32 + half * 16);
                an.q[1] = *(const uint4*)(erow + kn * 32 + half * 16 + 8);
                const uint4* bpn = wih_base + (size_t)kn * 64;
#pragma unroll
                for (int j = 0; j < 8; ++j) {
                    const uint4* p = bpn + (size_t)j * 512;
                    bn[j][0] = p[0]; bn[j][1] = p[1];
                }
#pragma unroll
                for (int j = 0; j < 8; ++j) {
                    AF b; b.q[0] = bb[j][0]; b.q[1] = bb[j][1];
                    acc[j] = __builtin_amdgcn_wmma_f32_16x16x32_f16(
                        false, a0.v, false, b.v, (short)0, acc[j], false, false);
                }
                a0 = an;
#pragma unroll
                for (int j = 0; j < 8; ++j) { bb[j][0] = bn[j][0]; bb[j][1] = bn[j][1]; }
            }
        }

        __syncthreads();                       /* all reads of h_old done */

        /* ---- h update: tanh + per-row length mask, write back f16 ---- */
#pragma unroll
        for (int j = 0; j < 8; ++j) {
            const int n  = (wave * 8 + j) * 16 + r16;
            const float bn = bias_lds[n];
#pragma unroll
            for (int i = 0; i < 8; ++i) {
                const int r   = i + half * 8;
                const int idx = r * HPAD + n;
                float hold = (float)h_lds[idx];
                float z    = acc[j][i] + bn;
                float nh   = tanhf(z);
                float hv   = (t < slen[r]) ? nh : hold;
                h_lds[idx] = (_Float16)hv;
            }
        }

        __syncthreads();                       /* h_new visible */

        /* ---- logits: h_new @ W_out^T (zero-padded 16x16 tile), K split over waves ---- */
        {
            uint4 ob[4][2];
#pragma unroll
            for (int k4 = 0; k4 < 4; ++k4) {
                const uint4* p = wout_base + (size_t)k4 * 64;
                ob[k4][0] = p[0]; ob[k4][1] = p[1];
            }
            v8f oc = {};
#pragma unroll
            for (int k4 = 0; k4 < 4; ++k4) {
                const int kk = wave * 4 + k4;
                AF a;
                loadA(a, abase, kk, half);
                AF b; b.q[0] = ob[k4][0]; b.q[1] = ob[k4][1];
                oc = __builtin_amdgcn_wmma_f32_16x16x32_f16(
                    false, a.v, false, b.v, (short)0, oc, false, false);
            }
#pragma unroll
            for (int i = 0; i < 8; ++i) outscr[wave * 256 + i * 32 + lane] = oc[i];
        }

        __syncthreads();                       /* outscr complete */

        {
            const int col = tid & 15;
            if (col < TAGS) {
                const int i  = tid >> 5;
                const int hf = (tid >> 4) & 1;
                const int r  = i + hf * 8;
                float s = b_out[col];
#pragma unroll
                for (int w = 0; w < NWAVE; ++w) s += outscr[w * 256 + tid];
                out[((size_t)(b0 + r) * T_SZ + t) * TAGS + col] = s;
            }
        }
        /* next iteration's GEMM only *reads* h_lds / ids; the first write to
           shared state (h update) is behind the post-GEMM barrier, so no
           barrier is needed at the loop boundary. */
    }
}

/* ---------------- host entry ---------------- */

extern "C" void kernel_launch(void* const* d_in, const int* in_sizes, int n_in,
                              void* d_out, int out_size, void* d_ws, size_t ws_size,
                              hipStream_t stream) {
    const int*   gid   = (const int*)d_in[0];
    const float* emb   = (const float*)d_in[1];
    const float* W_ih  = (const float*)d_in[2];
    const float* W_hh  = (const float*)d_in[3];
    const float* b_ih  = (const float*)d_in[4];
    const float* b_hh  = (const float*)d_in[5];
    const float* W_out = (const float*)d_in[6];
    const float* b_out = (const float*)d_in[7];

    _Float16* ws16  = (_Float16*)d_ws;
    _Float16* emb16 = ws16 + EMB16_OFF;
    _Float16* whh   = ws16 + WHH_OFF;
    _Float16* wih   = ws16 + WIH_OFF;
    _Float16* wout  = ws16 + WOUT_OFF;

    prep_emb <<<(VOCAB * E_SZ + 255) / 256, 256, 0, stream>>>(emb,   emb16);
    prep_whh <<<(H_SZ * H_SZ  + 255) / 256, 256, 0, stream>>>(W_hh,  whh);
    prep_wih <<<(H_SZ * E_SZ  + 255) / 256, 256, 0, stream>>>(W_ih,  wih);
    prep_wout<<<(16 * H_SZ    + 255) / 256, 256, 0, stream>>>(W_out, wout);

    rnn_kernel<<<B_SZ / BT, TPB, 0, stream>>>(
        gid, emb16,
        (const uint4*)whh, (const uint4*)wih, (const uint4*)wout,
        b_ih, b_hh, b_out, (float*)d_out);
}
